// UniEncoder_71030169141561
// MI455X (gfx1250) — compile-verified, matrix operand order checked by reading, not run on hardware
//
#include <hip/hip_runtime.h>
#include <hip/hip_bf16.h>

// ---------------------------------------------------------------------------
// MI455X (gfx1250) implementation of the 5-layer WGIN encoder.
// All GEMMs run on v_wmma_f32_16x16x32_f16 (wave32, 16x16 C tile per wave).
// ---------------------------------------------------------------------------

typedef __attribute__((ext_vector_type(16))) _Float16 v16h;
typedef __attribute__((ext_vector_type(8)))  _Float16 v8h;
typedef __attribute__((ext_vector_type(8)))  float    v8f;

#define NPG    116
#define NG     512
#define EMB    300
#define NNODE  (NG * NPG)      // 59392 (divisible by 16)
#define KP_EMB 320             // EMB padded to multiple of 32
#define NPAD   304             // EMB padded to multiple of 16
#define KP_F   128             // 116 padded to multiple of 32
#define BN_EPS 1e-5f

// Load one 16-lane-wide half fragment (A: 16xK row-major / B: stored K-major as
// Bt[N][K]).  CDNA5 16-bit layout: lane&15 = row (M or N), lane>>4 = K half,
// element e maps to K = (e<8?0:16) + 8*khalf + (e&7)  -> two contiguous 16B
// chunks per lane.
__device__ __forceinline__ v16h frag_load(const _Float16* __restrict__ base,
                                          int ld, int row, int k0, int khalf) {
  const _Float16* p = base + (size_t)row * ld + k0 + khalf * 8;
  v8h lo = *(const v8h*)(p);
  v8h hi = *(const v8h*)(p + 16);
  v16h f;
#pragma unroll
  for (int e = 0; e < 8; ++e) { f[e] = lo[e]; f[e + 8] = hi[e]; }
  return f;
}

// ---------------- adjacency construction (transposed: At[g][c][r]) ----------
__global__ void scatter_adj_i(const int* __restrict__ ei,
                              const float* __restrict__ w,
                              float* __restrict__ At, int E) {
  int e = blockIdx.x * blockDim.x + threadIdx.x;
  if (e >= E) return;
  int r = ei[e], c = ei[E + e];
  int g = r / NPG;
  atomicAdd(&At[((size_t)g * NPG + (c % NPG)) * NPG + (r % NPG)], w[e]);
}

__global__ void scatter_adj_f(const float* __restrict__ baw,
                              float* __restrict__ At, int E) {
  int e = blockIdx.x * blockDim.x + threadIdx.x;
  if (e >= E) return;
  int r = (int)baw[e];
  int c = (int)baw[(size_t)E + e];
  float w = baw[2 * (size_t)E + e];
  int g = r / NPG;
  atomicAdd(&At[((size_t)g * NPG + (c % NPG)) * NPG + (r % NPG)], w);
}

__global__ void adj_product(const float* __restrict__ A1,
                            const float* __restrict__ A2,
                            _Float16* __restrict__ At16) {
  int idx = blockIdx.x * blockDim.x + threadIdx.x;   // over NG*116*116
  if (idx >= NG * NPG * NPG) return;
  int g = idx / (NPG * NPG);
  int rem = idx - g * NPG * NPG;
  int c = rem / NPG, r = rem - c * NPG;
  At16[((size_t)g * KP_F + c) * KP_F + r] = (_Float16)(A1[idx] * A2[idx]);
}

// ---------------- per-layer transpose/convert of h: ht16[g][f][r] -----------
__global__ void transpose_h(const float* __restrict__ h, int F,
                            _Float16* __restrict__ ht) {
  int idx = blockIdx.x * blockDim.x + threadIdx.x;   // over NG*NPG*F
  if (idx >= NG * NPG * F) return;
  int g = idx / (NPG * F);
  int rem = idx - g * NPG * F;
  int r = rem / F;
  int f = rem - r * F;
  ht[((size_t)g * NPAD + f) * KP_F + r] = (_Float16)h[idx];
}

// ---------------- weight transpose/convert: Wt[n][k] f16 --------------------
__global__ void conv_w(const float* __restrict__ W, _Float16* __restrict__ Wt,
                       int K, int KP) {
  int idx = blockIdx.x * blockDim.x + threadIdx.x;   // over NPAD*KP
  if (idx >= NPAD * KP) return;
  int n = idx / KP, k = idx - n * KP;
  Wt[idx] = (n < EMB && k < K) ? (_Float16)W[(size_t)k * EMB + n]
                               : (_Float16)0.0f;
}

// ---------------- GEMM 1: z = h + At(g) @ h, f16 out ------------------------
// grid = NG*NT blocks of 256 (8 waves); wave w -> M tile w (M padded to 128).
__global__ void gemm_adj(const _Float16* __restrict__ At16,
                         const _Float16* __restrict__ ht16,
                         const float* __restrict__ hres, int F,
                         _Float16* __restrict__ z16, int NT) {
  int blk = blockIdx.x;
  int g   = blk / NT;
  int tn  = blk - g * NT;
  int wav = threadIdx.x >> 5;            // 0..7  -> M tile
  int lane = threadIdx.x & 31;
  int lrow = lane & 15, khalf = lane >> 4;
  const _Float16* A = At16 + (size_t)g * KP_F * KP_F;
  const _Float16* B = ht16 + (size_t)g * NPAD * KP_F;
  v8f acc = {};
#pragma unroll
  for (int k0 = 0; k0 < KP_F; k0 += 32) {
    v16h a = frag_load(A, KP_F, wav * 16 + lrow, k0, khalf);
    v16h b = frag_load(B, KP_F, tn * 16 + lrow, k0, khalf);
    acc = __builtin_amdgcn_wmma_f32_16x16x32_f16(false, a, false, b,
                                                 (short)0, acc, false, false);
  }
  int n = tn * 16 + lrow;
#pragma unroll
  for (int v = 0; v < 8; ++v) {
    int c = wav * 16 + v + 8 * khalf;    // C layout: VGPR v -> M = v + 8*khalf
    if (c < NPG) {
      size_t node = (size_t)g * NPG + c;
      float resid = (n < F) ? hres[node * (size_t)F + n] : 0.0f;
      z16[node * KP_EMB + n] = (_Float16)(resid + acc[v]);
    }
  }
}

// ---------------- GEMM 2/3: MLP layers --------------------------------------
// tiles: M = NNODE/16 = 3712, N = NPAD/16 = 19;  grid = 3712*19/8 = 8816.
__global__ void gemm_mlp(const _Float16* __restrict__ Amat,
                         const _Float16* __restrict__ Bt, int ldb, int K,
                         const float* __restrict__ bias,
                         _Float16* __restrict__ out16,
                         float* __restrict__ out32, int do_relu) {
  const int NTILES = NPAD / 16;          // 19
  int tile = blockIdx.x * 8 + (threadIdx.x >> 5);
  int tm = tile / NTILES;
  int tn = tile - tm * NTILES;
  int lane = threadIdx.x & 31;
  int lrow = lane & 15, khalf = lane >> 4;
  v8f acc = {};
  for (int k0 = 0; k0 < K; k0 += 32) {
    v16h a = frag_load(Amat, KP_EMB, tm * 16 + lrow, k0, khalf);
    v16h b = frag_load(Bt, ldb, tn * 16 + lrow, k0, khalf);
    acc = __builtin_amdgcn_wmma_f32_16x16x32_f16(false, a, false, b,
                                                 (short)0, acc, false, false);
  }
  int n = tn * 16 + lrow;
  float bv = (n < EMB) ? bias[n] : 0.0f;
#pragma unroll
  for (int v = 0; v < 8; ++v) {
    int row = tm * 16 + v + 8 * khalf;   // always < NNODE
    float val = acc[v] + bv;
    if (out16) {
      if (do_relu) val = fmaxf(val, 0.0f);
      out16[(size_t)row * KP_EMB + n] = (_Float16)val;
    } else if (n < EMB) {
      out32[(size_t)row * EMB + n] = val;
    }
  }
}

// ---------------- batchnorm ---------------------------------------------------
__global__ void bn_stats(const float* __restrict__ z, float* __restrict__ sum,
                         float* __restrict__ sumsq) {
  int c = threadIdx.x;                   // 320 threads, 300 active
  if (c >= EMB) return;
  size_t row0 = (size_t)blockIdx.x * 256;
  float s = 0.0f, q = 0.0f;
  for (int r = 0; r < 256; ++r) {
    float v = z[(row0 + r) * EMB + c];   // consecutive threads -> contiguous
    s += v; q += v * v;
  }
  atomicAdd(&sum[c], s);
  atomicAdd(&sumsq[c], q);
}

__global__ void bn_apply(const float* __restrict__ z,
                         const float* __restrict__ sum,
                         const float* __restrict__ sumsq,
                         const float* __restrict__ gamma,
                         const float* __restrict__ beta,
                         float* __restrict__ out, int do_relu) {
  size_t idx = (size_t)blockIdx.x * blockDim.x + threadIdx.x;
  if (idx >= (size_t)NNODE * EMB) return;
  int c = (int)(idx % EMB);
  const float invN = 1.0f / (float)NNODE;
  float mean = sum[c] * invN;
  float var  = sumsq[c] * invN - mean * mean;   // biased var, matches jnp.var
  float val = (z[idx] - mean) * rsqrtf(var + BN_EPS) * gamma[c] + beta[c];
  if (do_relu) val = fmaxf(val, 0.0f);
  out[idx] = val;
}

// ---------------- graph pooling ----------------------------------------------
__global__ void pool_sum(const float* __restrict__ xout,
                         float* __restrict__ xpool) {
  int g = blockIdx.x;
  int c = threadIdx.x;                   // 320 threads, 300 active
  if (c >= EMB) return;
  float s = 0.0f;
  for (int r = 0; r < NPG; ++r)
    s += xout[((size_t)g * NPG + r) * EMB + c];
  xpool[(size_t)g * EMB + c] = s;
}

// ---------------------------------------------------------------------------
extern "C" void kernel_launch(void* const* d_in, const int* in_sizes, int n_in,
                              void* d_out, int out_size, void* d_ws,
                              size_t ws_size, hipStream_t stream) {
  (void)n_in; (void)out_size; (void)ws_size;
  const float* x      = (const float*)d_in[1];
  const int*   ei     = (const int*)d_in[2];
  const float* ew     = (const float*)d_in[4];
  const float* baw    = (const float*)d_in[5];
  const float* W1_0   = (const float*)d_in[6];
  const float* b1_0   = (const float*)d_in[7];
  const float* W2_0   = (const float*)d_in[8];
  const float* b2_0   = (const float*)d_in[9];
  const float* W1s    = (const float*)d_in[10];
  const float* b1s    = (const float*)d_in[11];
  const float* W2s    = (const float*)d_in[12];
  const float* b2s    = (const float*)d_in[13];
  const float* gammas = (const float*)d_in[14];
  const float* betas  = (const float*)d_in[15];
  const int E = in_sizes[4];

  // ---- workspace carve-up (256B aligned) ----
  char* ws = (char*)d_ws;
  size_t off = 0;
  auto carve = [&](size_t bytes) -> void* {
    void* p = ws + off;
    off = (off + bytes + 255) & ~(size_t)255;
    return p;
  };
  float*    A1t   = (float*)   carve((size_t)NG * NPG * NPG * 4);
  float*    A2t   = (float*)   carve((size_t)NG * NPG * NPG * 4);
  _Float16* At16  = (_Float16*)carve((size_t)NG * KP_F * KP_F * 2);
  _Float16* ht16  = (_Float16*)carve((size_t)NG * NPAD * KP_F * 2);
  _Float16* z16   = (_Float16*)carve((size_t)NNODE * KP_EMB * 2);
  _Float16* t16   = (_Float16*)carve((size_t)NNODE * KP_EMB * 2);
  float*    z2    = (float*)   carve((size_t)NNODE * EMB * 4);
  float*    h32   = (float*)   carve((size_t)NNODE * EMB * 4);
  _Float16* W1_0t = (_Float16*)carve((size_t)NPAD * KP_F * 2);
  _Float16* Wbig  = (_Float16*)carve((size_t)9 * NPAD * KP_EMB * 2);
  float*    bnSum = (float*)   carve(NPAD * 4);
  float*    bnSq  = (float*)   carve(NPAD * 4);

  // zero accumulation buffers + all zero-padded f16 operand buffers
  hipMemsetAsync(A1t, 0, (size_t)NG * NPG * NPG * 4, stream);
  hipMemsetAsync(A2t, 0, (size_t)NG * NPG * NPG * 4, stream);
  hipMemsetAsync(At16, 0, (size_t)NG * KP_F * KP_F * 2, stream);
  hipMemsetAsync(ht16, 0, (size_t)NG * NPAD * KP_F * 2, stream);
  hipMemsetAsync(z16, 0, (size_t)NNODE * KP_EMB * 2, stream);
  hipMemsetAsync(t16, 0, (size_t)NNODE * KP_EMB * 2, stream);

  // adjacency: scatter both edge sets (transposed), multiply, convert to f16
  scatter_adj_i<<<(E + 255) / 256, 256, 0, stream>>>(ei, ew, A1t, E);
  scatter_adj_f<<<(E + 255) / 256, 256, 0, stream>>>(baw, A2t, E);
  adj_product<<<(NG * NPG * NPG + 255) / 256, 256, 0, stream>>>(A1t, A2t, At16);

  // weights -> f16, K-major, padded
  conv_w<<<(NPAD * KP_F + 255) / 256, 256, 0, stream>>>(W1_0, W1_0t, NPG, KP_F);
  _Float16* W2_0t = Wbig;
  conv_w<<<(NPAD * KP_EMB + 255) / 256, 256, 0, stream>>>(W2_0, W2_0t, EMB, KP_EMB);
  for (int l = 0; l < 4; ++l) {
    conv_w<<<(NPAD * KP_EMB + 255) / 256, 256, 0, stream>>>(
        W1s + (size_t)l * EMB * EMB, Wbig + (size_t)(1 + l) * NPAD * KP_EMB,
        EMB, KP_EMB);
    conv_w<<<(NPAD * KP_EMB + 255) / 256, 256, 0, stream>>>(
        W2s + (size_t)l * EMB * EMB, Wbig + (size_t)(5 + l) * NPAD * KP_EMB,
        EMB, KP_EMB);
  }

  float* xpool = (float*)d_out;
  float* xout  = xpool + (size_t)NG * EMB;

  for (int i = 0; i < 5; ++i) {
    const int F  = (i == 0) ? NPG : EMB;
    const int NT = (i == 0) ? 8 : 19;            // N tiles for adjacency GEMM
    const int K1 = (i == 0) ? KP_F : KP_EMB;     // K for MLP GEMM 1
    const float* hsrc = (i == 0) ? x : h32;

    transpose_h<<<(NG * NPG * F + 255) / 256, 256, 0, stream>>>(hsrc, F, ht16);
    gemm_adj<<<NG * NT, 256, 0, stream>>>(At16, ht16, hsrc, F, z16, NT);

    const _Float16* W1t = (i == 0) ? W1_0t
                                   : Wbig + (size_t)(1 + (i - 1)) * NPAD * KP_EMB;
    const int ldb1 = (i == 0) ? KP_F : KP_EMB;
    const float* b1 = (i == 0) ? b1_0 : b1s + (size_t)(i - 1) * EMB;
    const _Float16* W2t = (i == 0) ? W2_0t
                                   : Wbig + (size_t)(5 + (i - 1)) * NPAD * KP_EMB;
    const float* b2 = (i == 0) ? b2_0 : b2s + (size_t)(i - 1) * EMB;

    gemm_mlp<<<(NNODE / 16) * (NPAD / 16) / 8, 256, 0, stream>>>(
        z16, W1t, ldb1, K1, b1, t16, nullptr, 1);
    gemm_mlp<<<(NNODE / 16) * (NPAD / 16) / 8, 256, 0, stream>>>(
        t16, W2t, KP_EMB, KP_EMB, b2, nullptr, z2, 0);

    hipMemsetAsync(bnSum, 0, NPAD * 4, stream);
    hipMemsetAsync(bnSq, 0, NPAD * 4, stream);
    bn_stats<<<NNODE / 256, 320, 0, stream>>>(z2, bnSum, bnSq);
    float* dst = (i < 4) ? h32 : xout;
    bn_apply<<<(int)(((size_t)NNODE * EMB + 255) / 256), 256, 0, stream>>>(
        z2, bnSum, bnSq, gammas + (size_t)i * EMB, betas + (size_t)i * EMB,
        dst, (i < 4) ? 1 : 0);
  }

  pool_sum<<<NG, 320, 0, stream>>>(xout, xpool);
}
